// GraphConvolution_72567767433676
// MI455X (gfx1250) — compile-verified, matrix operand order checked by reading, not run on hardware
//
#include <hip/hip_runtime.h>
#include <hip/hip_bf16.h>

// Problem constants (from reference)
#define NDIM 4096
#define FDIM 256
#define KKER 4
#define NNZE 65536

typedef unsigned short u16;
typedef __attribute__((ext_vector_type(16))) __bf16 v16bf;
typedef __attribute__((ext_vector_type(8)))  float  v8f;

// Fragment register image: 8 dwords = 16 bf16, loadable as two uint4.
union FragBF { uint4 q[2]; unsigned u[8]; v16bf v; };

__device__ __forceinline__ unsigned bf16_rne_bits(float f) {
  unsigned u = __float_as_uint(f);
  unsigned lsb = (u >> 16) & 1u;
  return (u + 0x7FFFu + lsb) >> 16;
}
__device__ __forceinline__ float bf16_bits_to_f32(unsigned b) {
  return __uint_as_float(b << 16);
}
__device__ __forceinline__ void split_hi_lo(float f, unsigned& h, unsigned& l) {
  h = bf16_rne_bits(f);
  l = bf16_rne_bits(f - bf16_bits_to_f32(h));
}

// ---------------------------------------------------------------------------
// Pass 1: split f32 rows into hi/lo bf16 planes (same layout). Run once for X.
// ---------------------------------------------------------------------------
__global__ __launch_bounds__(256)
void split_bf16_rows(const float* __restrict__ src, u16* __restrict__ hi,
                     u16* __restrict__ lo) {
  const size_t idx = ((size_t)blockIdx.x * 256 + threadIdx.x) * 4;
  float4 v = *(const float4*)(src + idx);
  unsigned h0, h1, h2, h3, l0, l1, l2, l3;
  split_hi_lo(v.x, h0, l0);
  split_hi_lo(v.y, h1, l1);
  split_hi_lo(v.z, h2, l2);
  split_hi_lo(v.w, h3, l3);
  uint2 ph, pl;
  ph.x = h0 | (h1 << 16); ph.y = h2 | (h3 << 16);
  pl.x = l0 | (l1 << 16); pl.y = l2 | (l3 << 16);
  *(uint2*)(hi + idx) = ph;
  *(uint2*)(lo + idx) = pl;
}

// ---------------------------------------------------------------------------
// Pass 2 (per k): W[FDIM][NDIM] f32 -> WT hi/lo bf16 planes [NDIM][FDIM].
// Pre-transposing makes GEMM's B staging identical to A (K-contiguous rows).
// ---------------------------------------------------------------------------
__global__ __launch_bounds__(256)
void split_bf16_transpose(const float* __restrict__ W, u16* __restrict__ hiT,
                          u16* __restrict__ loT) {
  __shared__ float t[32][33];
  const int j0 = blockIdx.x * 32;  // NDIM direction
  const int f0 = blockIdx.y * 32;  // FDIM direction
  const int c  = threadIdx.x & 31;
  const int r0 = threadIdx.x >> 5;  // 0..7
#pragma unroll
  for (int uu = 0; uu < 4; ++uu) {
    int r = r0 + uu * 8;
    t[r][c] = W[(size_t)(f0 + r) * NDIM + j0 + c];
  }
  __syncthreads();
#pragma unroll
  for (int uu = 0; uu < 4; ++uu) {
    int r = r0 + uu * 8;
    float f = t[c][r];  // = W[f0+c][j0+r]
    unsigned h, l;
    split_hi_lo(f, h, l);
    hiT[(size_t)(j0 + r) * FDIM + f0 + c] = (u16)h;
    loT[(size_t)(j0 + r) * FDIM + f0 + c] = (u16)l;
  }
}

// ---------------------------------------------------------------------------
// GEMM: Yt = (X @ W)^T with split-bf16 (hi*hi + hi*lo + lo*hi) WMMA.
// Block tile 128(M) x 128(N), 8 waves, wave tile 32x64 (2x4 WMMA accs).
// All operands arrive as bf16 planes; fragment build = 2x ds_load_b128, 0 VALU.
// ---------------------------------------------------------------------------
#define LDPH 40  // LDS row stride in bf16 units (80B): 16B-aligned, conflict-free

__global__ __launch_bounds__(256)
void gemm_bf16x3_yt(const u16* __restrict__ Ahi, const u16* __restrict__ Alo,
                    const u16* __restrict__ Bhi, const u16* __restrict__ Blo,
                    float* __restrict__ Yt) {
  __shared__ u16 AH[128][LDPH];
  __shared__ u16 AL[128][LDPH];
  __shared__ u16 BH[128][LDPH];
  __shared__ u16 BL[128][LDPH];
  const int tid  = threadIdx.x;
  const int lane = tid & 31;
  const int w    = tid >> 5;
  const int wm   = w & 3;   // M offset 32*wm
  const int wn   = w >> 2;  // N offset 64*wn
  const int i0   = blockIdx.y * 128;
  const int j0   = blockIdx.x * 128;
  const int h    = (lane >> 4) & 1;
  const int rsel = lane & 15;

  v8f acc[2][4];
  {
    v8f vz = {};
#pragma unroll
    for (int m = 0; m < 2; ++m)
#pragma unroll
      for (int n = 0; n < 4; ++n) acc[m][n] = vz;
  }

  for (int kt = 0; kt < FDIM; kt += 32) {
    // ---- Stage 4 bf16 tiles [128 rows][32 k]: 512 x 16B chunks per tile.
#pragma unroll
    for (int it = 0; it < 2; ++it) {
      int q   = tid + 256 * it;
      int row = q >> 2;
      int qc  = q & 3;
      size_t goA = (size_t)(i0 + row) * FDIM + kt + qc * 8;
      size_t goB = (size_t)(j0 + row) * FDIM + kt + qc * 8;
      *(uint4*)&AH[row][qc * 8] = *(const uint4*)(Ahi + goA);
      *(uint4*)&AL[row][qc * 8] = *(const uint4*)(Alo + goA);
      *(uint4*)&BH[row][qc * 8] = *(const uint4*)(Bhi + goB);
      *(uint4*)&BL[row][qc * 8] = *(const uint4*)(Blo + goB);
    }
    // Prefetch next K-slice (global_prefetch_b8)
    if (kt + 32 < FDIM) {
      __builtin_prefetch(Ahi + (size_t)(i0 + (tid >> 1)) * FDIM + kt + 32, 0, 1);
      __builtin_prefetch(Bhi + (size_t)(j0 + (tid >> 1)) * FDIM + kt + 32, 0, 1);
    }
    __syncthreads();

    // ---- Fragment build: ISA 16-bit A layout -> dwords [g*4+d] hold
    //      K = g*16 + h*8 + 2d..2d+1, i.e. 8 contiguous bf16 (16B) per group.
    FragBF aHi[2], aLo[2], bHi[4], bLo[4];
#pragma unroll
    for (int m = 0; m < 2; ++m) {
      const u16* pH = &AH[wm * 32 + m * 16 + rsel][h * 8];
      const u16* pL = &AL[wm * 32 + m * 16 + rsel][h * 8];
      aHi[m].q[0] = *(const uint4*)pH;
      aHi[m].q[1] = *(const uint4*)(pH + 16);
      aLo[m].q[0] = *(const uint4*)pL;
      aLo[m].q[1] = *(const uint4*)(pL + 16);
    }
#pragma unroll
    for (int n = 0; n < 4; ++n) {
      const u16* pH = &BH[wn * 64 + n * 16 + rsel][h * 8];
      const u16* pL = &BL[wn * 64 + n * 16 + rsel][h * 8];
      bHi[n].q[0] = *(const uint4*)pH;
      bHi[n].q[1] = *(const uint4*)(pH + 16);
      bLo[n].q[0] = *(const uint4*)pL;
      bLo[n].q[1] = *(const uint4*)(pL + 16);
    }

#pragma unroll
    for (int m = 0; m < 2; ++m)
#pragma unroll
      for (int n = 0; n < 4; ++n) {
        acc[m][n] = __builtin_amdgcn_wmma_f32_16x16x32_bf16(
            false, aHi[m].v, false, bHi[n].v, (short)0, acc[m][n], false, false);
        acc[m][n] = __builtin_amdgcn_wmma_f32_16x16x32_bf16(
            false, aLo[m].v, false, bHi[n].v, (short)0, acc[m][n], false, false);
        acc[m][n] = __builtin_amdgcn_wmma_f32_16x16x32_bf16(
            false, aHi[m].v, false, bLo[n].v, (short)0, acc[m][n], false, false);
      }
    __syncthreads();
  }

  // ---- Store transposed: D layout gives 8 consecutive M per lane -> 2x b128
#pragma unroll
  for (int m = 0; m < 2; ++m) {
    int ib = i0 + wm * 32 + m * 16 + (h << 3);
#pragma unroll
    for (int n = 0; n < 4; ++n) {
      int jj = j0 + wn * 64 + n * 16 + rsel;
      float* dst = Yt + (size_t)jj * NDIM + ib;
      v8f a = acc[m][n];
      *(float4*)dst       = make_float4(a[0], a[1], a[2], a[3]);
      *((float4*)dst + 1) = make_float4(a[4], a[5], a[6], a[7]);
    }
  }
}

// ---------------------------------------------------------------------------
// Scatter: outT[r, i] += v * Yt[c, i] per COO edge. Each block exclusively
// owns a 256-wide i-slice -> race-free, deterministic, no atomics. Yt (64MB)
// + outT (64MB) both stay resident in the 192MB L2.
// ---------------------------------------------------------------------------
__global__ __launch_bounds__(256)
void scatter_spmm(const float* __restrict__ Yt, const int* __restrict__ idx,
                  const float* __restrict__ vals, float* __restrict__ outT) {
  const int i = blockIdx.x * 256 + threadIdx.x;
#pragma unroll 4
  for (int e = 0; e < NNZE; ++e) {
    const int   r = idx[2 * e];  // uniform -> scalar load
    const int   c = idx[2 * e + 1];
    const float v = vals[e];
    float* o = outT + (size_t)r * NDIM + i;
    *o += v * Yt[(size_t)c * NDIM + i];
  }
}

// ---------------------------------------------------------------------------
// In-place: out[i][j] = relu(outT[j][i] + bias[j]) via 32x32 tile pairs.
// ---------------------------------------------------------------------------
__global__ __launch_bounds__(256)
void transpose_bias_relu_inplace(float* __restrict__ buf, const float* __restrict__ bias) {
  const int a = blockIdx.y;
  const int b = blockIdx.x;
  if (a > b) return;
  __shared__ float ta[32][33];
  __shared__ float tb[32][33];
  const int c  = threadIdx.x & 31;
  const int r0 = threadIdx.x >> 5;  // 0..7
#pragma unroll
  for (int uu = 0; uu < 4; ++uu) {
    int r = r0 + uu * 8;
    ta[r][c] = buf[(size_t)(a * 32 + r) * NDIM + b * 32 + c];
    tb[r][c] = buf[(size_t)(b * 32 + r) * NDIM + a * 32 + c];
  }
  __syncthreads();
  const float biasB = bias[b * 32 + c];
  const float biasA = bias[a * 32 + c];
#pragma unroll
  for (int uu = 0; uu < 4; ++uu) {
    int r = r0 + uu * 8;
    float v1 = tb[c][r] + biasB;
    buf[(size_t)(a * 32 + r) * NDIM + b * 32 + c] = v1 > 0.f ? v1 : 0.f;
    if (a != b) {
      float v2 = ta[c][r] + biasA;
      buf[(size_t)(b * 32 + r) * NDIM + a * 32 + c] = v2 > 0.f ? v2 : 0.f;
    }
  }
}

extern "C" void kernel_launch(void* const* d_in, const int* in_sizes, int n_in,
                              void* d_out, int out_size, void* d_ws, size_t ws_size,
                              hipStream_t stream) {
  (void)in_sizes; (void)n_in; (void)out_size; (void)ws_size;
  const float* X    = (const float*)d_in[0];  // (4096, 256)
  const float* Ker  = (const float*)d_in[1];  // (4, 256, 4096)
  const float* bias = (const float*)d_in[2];  // (4096,)
  const float* sv   = (const float*)d_in[3];  // (4, 65536)
  const int*   si   = (const int*)  d_in[4];  // (4, 65536, 2)
  float* outT = (float*)d_out;  // accumulates out^T, finalized in-place

  // Workspace layout (72 MB total):
  //   [0, 64MB)   Yt   : (X @ W_k)^T, f32
  //   [+0, +2MB)  Xhi  : bf16 hi plane of X        [NDIM? no: NDIM x FDIM]
  //   then Xlo, WThi, WTlo (2MB each)
  float* Yt = (float*)d_ws;
  u16* Xhi  = (u16*)((char*)d_ws + (size_t)NDIM * NDIM * sizeof(float));
  u16* Xlo  = Xhi  + (size_t)NDIM * FDIM;
  u16* WThi = Xlo  + (size_t)NDIM * FDIM;
  u16* WTlo = WThi + (size_t)NDIM * FDIM;

  hipMemsetAsync(d_out, 0, (size_t)NDIM * NDIM * sizeof(float), stream);

  // One-time split of X into hi/lo bf16 planes.
  split_bf16_rows<<<(NDIM * FDIM) / 1024, 256, 0, stream>>>(X, Xhi, Xlo);

  dim3 gGemm(NDIM / 128, NDIM / 128);
  dim3 gSplitW(NDIM / 32, FDIM / 32);
  for (int k = 0; k < KKER; ++k) {
    split_bf16_transpose<<<gSplitW, 256, 0, stream>>>(
        Ker + (size_t)k * FDIM * NDIM, WThi, WTlo);
    gemm_bf16x3_yt<<<gGemm, 256, 0, stream>>>(Xhi, Xlo, WThi, WTlo, Yt);
    scatter_spmm<<<NDIM / 256, 256, 0, stream>>>(
        Yt, si + (size_t)k * NNZE * 2, sv + (size_t)k * NNZE, outT);
  }
  dim3 gT(NDIM / 32, NDIM / 32);
  transpose_bias_relu_inplace<<<gT, 256, 0, stream>>>(outT, bias);
}
